// FlexAttnBlock_85744727097361
// MI455X (gfx1250) — matrix operand order repaired and checked by reading, not compiled, not run on hardware
//
#include <hip/hip_runtime.h>
#include <stdint.h>

// ---------------------------------------------------------------------------
// Types for CDNA5 WMMA (wave32): V_WMMA_F32_16X16X32_BF16
// ---------------------------------------------------------------------------
typedef __attribute__((ext_vector_type(16))) __bf16 v16bf;
typedef __attribute__((ext_vector_type(8)))  float  v8f;
typedef __attribute__((ext_vector_type(4)))  unsigned int u32x4;
typedef __attribute__((ext_vector_type(8)))  int    i32x8;
typedef __attribute__((ext_vector_type(4)))  int    i32x4;

union BF16Frag {
    v16bf v;
    uint4 q[2];
};

#define D_MODEL 768
#define HW      2304       // 48*48
#define NHEADS  12
#define DHEAD   64
#define SCALE   0.125f     // 64^-0.5
#define KSTEPS  24         // 768 / 32

// LDS B-tile: 64 rows x 64B data padded by TDM to 80B pitch (bank-conflict free,
// 16B aligned for ds_load_b128). 2592 ushorts = 5184B per buffer (>= 64*80+16).
#define LDSB_PITCH_U16 40      // 80 bytes
#define LDSB_BUF_U16   2592

// ---------------------------------------------------------------------------
// fp32 -> bf16 (round to nearest even)
// ---------------------------------------------------------------------------
__device__ __forceinline__ unsigned short f2bf(float f) {
    unsigned int u = __float_as_uint(f);
    u += 0x7FFFu + ((u >> 16) & 1u);
    return (unsigned short)(u >> 16);
}

// 16-lane-group reductions via ds_swizzle (xor 1,2,4,8 with and_mask=0x1f)
__device__ __forceinline__ float xmax16(float v) {
    v = fmaxf(v, __int_as_float(__builtin_amdgcn_ds_swizzle(__float_as_int(v), 0x041F)));
    v = fmaxf(v, __int_as_float(__builtin_amdgcn_ds_swizzle(__float_as_int(v), 0x081F)));
    v = fmaxf(v, __int_as_float(__builtin_amdgcn_ds_swizzle(__float_as_int(v), 0x101F)));
    v = fmaxf(v, __int_as_float(__builtin_amdgcn_ds_swizzle(__float_as_int(v), 0x201F)));
    return v;
}
__device__ __forceinline__ float xsum16(float v) {
    v += __int_as_float(__builtin_amdgcn_ds_swizzle(__float_as_int(v), 0x041F));
    v += __int_as_float(__builtin_amdgcn_ds_swizzle(__float_as_int(v), 0x081F));
    v += __int_as_float(__builtin_amdgcn_ds_swizzle(__float_as_int(v), 0x101F));
    v += __int_as_float(__builtin_amdgcn_ds_swizzle(__float_as_int(v), 0x201F));
    return v;
}

// ---------------------------------------------------------------------------
// Fragment loaders (layouts per CDNA5 ISA §7.12.2)
// A 16x32 bf16: lanes 0-15 -> M=lane, K {0..7,16..23}; lanes 16-31 -> {8..15,24..31}
// ---------------------------------------------------------------------------
__device__ __forceinline__ v16bf load_a_frag(const unsigned short* rowPtr, int hi) {
    BF16Frag t;
    t.q[0] = *reinterpret_cast<const uint4*>(rowPtr + hi * 8);
    t.q[1] = *reinterpret_cast<const uint4*>(rowPtr + 16 + hi * 8);
    return t.v;
}
// B 32x16 bf16: lanes 0-15 hold K {0..7,16..23} for column N=lane; lanes 16-31 hold {8..15,24..31}.
// colPtr points at the start of the (contiguous-in-K) column.
__device__ __forceinline__ v16bf load_b_frag(const unsigned short* colPtr, int hi) {
    BF16Frag t;
    t.q[0] = *reinterpret_cast<const uint4*>(colPtr + hi * 8);
    t.q[1] = *reinterpret_cast<const uint4*>(colPtr + 16 + hi * 8);
    return t.v;
}
__device__ __forceinline__ v8f wmma_bf16(v16bf a, v16bf b, v8f c) {
    return __builtin_amdgcn_wmma_f32_16x16x32_bf16(false, a, false, b, (short)0, c,
                                                   false, false);
}

// ---------------------------------------------------------------------------
// TDM: DMA a 64-row x 32-element bf16 tile (row stride 768 elems) into LDS,
// padding each 64B row to 80B pitch. Descriptor per CDNA5 ISA.
// Issued once per wave; tracked with TENSORcnt.
// ---------------------------------------------------------------------------
__device__ __forceinline__ void tdm_load_w_tile(unsigned lds_byte_off,
                                                const unsigned short* gsrc) {
    const unsigned long long ga = (unsigned long long)(uintptr_t)gsrc;
    u32x4 g0;
    g0[0] = 1u;                                   // count=1 valid descriptor
    g0[1] = lds_byte_off;                         // lds_addr
    g0[2] = (unsigned)(ga & 0xFFFFFFFFu);         // global_addr[31:0]
    g0[3] = (unsigned)((ga >> 32) & 0x01FFFFFFu)  // global_addr[56:32]
          | 0x80000000u;                          // type = 2 ("image")
    i32x8 g1;
    g1[0] = (1 << 16)        // data_size = 2 bytes
          | (1 << 20)        // pad_enable
          | (3 << 22)        // pad_interval: 16 dwords (= one 64B row)
          | (3 << 25);       // pad_amount: 4 dwords (row pitch 80B)
    g1[1] = (32 << 16);      // tensor_dim0[15:0]=32 (tile interior, no OOB)
    g1[2] = (64 << 16);      // tensor_dim0[31:16]=0 | tensor_dim1[15:0]=64
    g1[3] = (32 << 16);      // tensor_dim1[31:16]=0 | tile_dim0=32
    g1[4] = 64;              // tile_dim1=64, tile_dim2=0
    g1[5] = D_MODEL;         // tensor_dim0_stride[31:0] = 768
    g1[6] = 0;               // stride0[47:32]=0, stride1[15:0]=0
    g1[7] = 0;
    i32x4 z4 = {0, 0, 0, 0};                      // groups 2/3 unused (2D tile)
    i32x8 z8 = {0, 0, 0, 0, 0, 0, 0, 0};
    __builtin_amdgcn_tensor_load_to_lds(g0, g1, z4, z4, z8, 0);
}

// ---------------------------------------------------------------------------
// Shared GEMM panel: one block = 128(M) x 64(N) C-tile, 8 waves (16 rows each).
// Weight tile (64x32) double-buffered in LDS via TDM; A double-buffered in regs.
// acc[j] covers columns n0 + j*16 + (lane&15).
// ---------------------------------------------------------------------------
__device__ __forceinline__ void gemm_panel(const unsigned short* __restrict__ A,
                                           const unsigned short* __restrict__ W,
                                           unsigned short* ldsB,
                                           int mrow, int n0,
                                           int wv, int hi, int l15,
                                           v8f (&acc)[4]) {
    const unsigned short* arow = A + (mrow + l15) * D_MODEL;
    const bool issuer = (wv == 0);
    const unsigned lds0 = (unsigned)(uintptr_t)ldsB;

    if (issuer) tdm_load_w_tile(lds0, W + n0 * D_MODEL);

    BF16Frag aCur, aNxt;
    aCur.q[0] = *reinterpret_cast<const uint4*>(arow + hi * 8);
    aCur.q[1] = *reinterpret_cast<const uint4*>(arow + 16 + hi * 8);

    for (int i = 0; i < KSTEPS; ++i) {
        const int cur = i & 1;
        if (i + 1 < KSTEPS) {
            __syncthreads();                       // buf[cur^1] free to overwrite
            if (issuer) {
                tdm_load_w_tile(lds0 + (unsigned)((cur ^ 1) * LDSB_BUF_U16 * 2),
                                W + n0 * D_MODEL + (i + 1) * 32);
                __builtin_amdgcn_s_wait_tensorcnt((short)1);  // tile i landed
            }
        } else if (issuer) {
            __builtin_amdgcn_s_wait_tensorcnt((short)0);
        }
        __syncthreads();                           // buf[cur] visible to all waves

        if (i + 1 < KSTEPS) {                      // prefetch next A chunk (regs)
            const int k = (i + 1) * 32;
            aNxt.q[0] = *reinterpret_cast<const uint4*>(arow + k + hi * 8);
            aNxt.q[1] = *reinterpret_cast<const uint4*>(arow + k + 16 + hi * 8);
        }
        const unsigned short* lb = ldsB + cur * LDSB_BUF_U16;
        acc[0] = wmma_bf16(aCur.v, load_b_frag(lb + (l15)      * LDSB_PITCH_U16, hi), acc[0]);
        acc[1] = wmma_bf16(aCur.v, load_b_frag(lb + (16 + l15) * LDSB_PITCH_U16, hi), acc[1]);
        acc[2] = wmma_bf16(aCur.v, load_b_frag(lb + (32 + l15) * LDSB_PITCH_U16, hi), acc[2]);
        acc[3] = wmma_bf16(aCur.v, load_b_frag(lb + (48 + l15) * LDSB_PITCH_U16, hi), acc[3]);
        aCur = aNxt;
    }
}

// ---------------------------------------------------------------------------
// Kernel 1: fp32 -> bf16 convert (4 elements / thread)
// ---------------------------------------------------------------------------
__global__ void __launch_bounds__(256) k_cvt(const float* __restrict__ src,
                                             unsigned short* __restrict__ dst, int n4) {
    int i = blockIdx.x * blockDim.x + threadIdx.x;
    if (i >= n4) return;
    float4 f = reinterpret_cast<const float4*>(src)[i];
    uint2 o;
    o.x = (unsigned)f2bf(f.x) | ((unsigned)f2bf(f.y) << 16);
    o.y = (unsigned)f2bf(f.z) | ((unsigned)f2bf(f.w) << 16);
    reinterpret_cast<uint2*>(dst)[i] = o;
}

// ---------------------------------------------------------------------------
// Kernel 2: QKV GEMM. C[4608 x 2304] = X[4608x768] * Wq[2304x768]^T
// Block = 128x64 tile. Q,K -> QK[4608 x 1536]; V -> Vt[b][head][e][hw].
// ---------------------------------------------------------------------------
__global__ void __launch_bounds__(256) k_gemm_qkv(const unsigned short* __restrict__ X,
                                                  const unsigned short* __restrict__ Wq,
                                                  unsigned short* __restrict__ QK,
                                                  unsigned short* __restrict__ Vt) {
    __shared__ __align__(16) unsigned short ldsB[2 * LDSB_BUF_U16];
    const int lane = threadIdx.x & 31;
    const int wv = threadIdx.x >> 5;
    const int hi = lane >> 4, l15 = lane & 15;
    const int mt = blockIdx.x / 36, nt = blockIdx.x % 36;   // 36 x 36 blocks
    const int mrow = mt * 128 + wv * 16;
    const int n0 = nt * 64;

    v8f acc[4] = {};
    gemm_panel(X, Wq, ldsB, mrow, n0, wv, hi, l15, acc);

#pragma unroll
    for (int j = 0; j < 4; ++j) {
        const int f = n0 + j * 16 + l15;
#pragma unroll
        for (int v = 0; v < 8; ++v) {
            const int r = mrow + v + 8 * hi;
            const unsigned short h = f2bf(acc[j][v]);
            if (f < 1536) {
                QK[r * 1536 + f] = h;                       // Q (f<768) and K rows
            } else {
                const int bb = r / HW, q = r % HW;
                const int fe = f - 1536;                    // head*64 + e
                Vt[(bb * 768 + fe) * HW + q] = h;           // V transposed
            }
        }
    }
}

// ---------------------------------------------------------------------------
// Kernel 3: neighborhood flash attention. One wave per (b, head, 16-query tile).
// 16 | 48 => each q tile has constant qx and a contiguous 16-wide qy range.
// ---------------------------------------------------------------------------
__global__ void __launch_bounds__(128) k_attn(const unsigned short* __restrict__ QK,
                                              const unsigned short* __restrict__ Vt,
                                              unsigned short* __restrict__ O,
                                              const int* __restrict__ ksz_ptr) {
    __shared__ __align__(16) unsigned short pTile[4][16 * 32];   // P staging per wave
    const int KS   = *ksz_ptr;
    const int wv   = threadIdx.x >> 5;
    const int wid  = blockIdx.x * 4 + wv;                        // 3456 tasks, exact
    const int lane = threadIdx.x & 31;
    const int hi = lane >> 4, l15 = lane & 15;

    const int b    = wid / (NHEADS * 144);
    const int rr   = wid % (NHEADS * 144);
    const int head = rr / 144;
    const int q0   = (rr % 144) * 16;
    const int qx   = q0 / 48;
    const int qy0  = q0 % 48;

    const unsigned short* qrow = QK + (b * HW + q0 + l15) * 1536 + head * DHEAD;
    const v16bf qf0 = load_a_frag(qrow, hi);
    const v16bf qf1 = load_a_frag(qrow + 32, hi);

    v8f o0 = {}, o1 = {}, o2 = {}, o3 = {};
    float rmax[8], rsum[8];
#pragma unroll
    for (int v = 0; v < 8; ++v) { rmax[v] = -__builtin_inff(); rsum[v] = 0.f; }

    unsigned short* pT = pTile[wv];
    const unsigned short* Kbase = QK + b * HW * 1536 + D_MODEL + head * DHEAD;
    const unsigned short* Vbase = Vt + (b * 768 + head * DHEAD) * HW;

    for (int k0 = 0; k0 < HW; k0 += 32) {
        // scalar tile culling on both 16-key halves
        const int kxa = k0 / 48,        kya = k0 % 48;
        const int kxb = (k0 + 16) / 48, kyb = (k0 + 16) % 48;
        const bool run_a = (qx - kxa <= KS) && (kxa - qx <= KS) &&
                           (qy0 - kya <= KS + 15) && (kya - qy0 <= KS + 15);
        const bool run_b = (qx - kxb <= KS) && (kxb - qx <= KS) &&
                           (qy0 - kyb <= KS + 15) && (kyb - qy0 <= KS + 15);
        if (!(run_a || run_b)) continue;

        // S = Q * K^T for 32 keys (two 16-column halves)
        v8f s0 = {}, s1 = {};
        const unsigned short* c0 = Kbase + (k0 + l15) * 1536;
        const unsigned short* c1 = Kbase + (k0 + 16 + l15) * 1536;
        s0 = wmma_bf16(qf0, load_b_frag(c0,      hi), s0);
        s0 = wmma_bf16(qf1, load_b_frag(c0 + 32, hi), s0);
        s1 = wmma_bf16(qf0, load_b_frag(c1,      hi), s1);
        s1 = wmma_bf16(qf1, load_b_frag(c1 + 32, hi), s1);

        // mask + online softmax (rows v+8*hi live in this lane's 16-lane group)
        const int myy = qy0 + 8 * hi;
        const int kyA = kya + l15, kyB = kyb + l15;
#pragma unroll
        for (int v = 0; v < 8; ++v) {
            const int dyA = myy + v - kyA;
            const int dyB = myy + v - kyB;
            const bool okA = run_a && (dyA <= KS) && (-dyA <= KS);
            const bool okB = run_b && (dyB <= KS) && (-dyB <= KS);
            const float sa = s0[v] * SCALE;
            const float sb = s1[v] * SCALE;
            const float ma = okA ? sa : -__builtin_inff();
            const float mb = okB ? sb : -__builtin_inff();
            const float tmax = xmax16(fmaxf(ma, mb));
            const float mnew = fmaxf(rmax[v], tmax);
            // fminf(NaN,0)=0 -> corr=1 for still-all-masked rows (acc stays 0)
            const float corr = __expf(fminf(rmax[v] - mnew, 0.f));
            const float pa = okA ? __expf(sa - mnew) : 0.f;
            const float pb = okB ? __expf(sb - mnew) : 0.f;
            rsum[v] = rsum[v] * corr + xsum16(pa + pb);
            rmax[v] = mnew;
            const int m = v + 8 * hi;
            pT[m * 32 + l15]      = f2bf(pa);
            pT[m * 32 + 16 + l15] = f2bf(pb);
            o0[v] *= corr; o1[v] *= corr; o2[v] *= corr; o3[v] *= corr;
        }

        // restage P as an A fragment (per-wave LDS; DS is in-order per wave)
        asm volatile("" ::: "memory");
        __builtin_amdgcn_wave_barrier();
        asm volatile("s_wait_dscnt 0" ::: "memory");
        BF16Frag pf;
        pf.q[0] = *reinterpret_cast<const uint4*>(pT + l15 * 32 + hi * 8);
        pf.q[1] = *reinterpret_cast<const uint4*>(pT + l15 * 32 + 16 + hi * 8);
        asm volatile("" ::: "memory");

        // O += P * V  (V transposed: columns contiguous along keys)
        const unsigned short* vc = Vbase + l15 * HW + k0;
        o0 = wmma_bf16(pf.v, load_b_frag(vc,           hi), o0);
        o1 = wmma_bf16(pf.v, load_b_frag(vc + 16 * HW, hi), o1);
        o2 = wmma_bf16(pf.v, load_b_frag(vc + 32 * HW, hi), o2);
        o3 = wmma_bf16(pf.v, load_b_frag(vc + 48 * HW, hi), o3);
        __builtin_amdgcn_wave_barrier();
    }

    // normalize and emit O[b*HW+q][head*64+e] in bf16 (self-key always unmasked)
    unsigned short* orow = O + (b * HW + q0) * D_MODEL + head * DHEAD + l15;
#pragma unroll
    for (int v = 0; v < 8; ++v) {
        const float inv = 1.0f / rsum[v];
        unsigned short* p = orow + (v + 8 * hi) * D_MODEL;
        p[0]  = f2bf(o0[v] * inv);
        p[16] = f2bf(o1[v] * inv);
        p[32] = f2bf(o2[v] * inv);
        p[48] = f2bf(o3[v] * inv);
    }
}

// ---------------------------------------------------------------------------
// Kernel 4: out = O[4608x768] * Wo[768x768]^T  (fp32 result)
// ---------------------------------------------------------------------------
__global__ void __launch_bounds__(256) k_gemm_out(const unsigned short* __restrict__ Oi,
                                                  const unsigned short* __restrict__ Wo,
                                                  float* __restrict__ out) {
    __shared__ __align__(16) unsigned short ldsB[2 * LDSB_BUF_U16];
    const int lane = threadIdx.x & 31;
    const int wv = threadIdx.x >> 5;
    const int hi = lane >> 4, l15 = lane & 15;
    const int mt = blockIdx.x / 12, nt = blockIdx.x % 12;   // 36 x 12 blocks
    const int mrow = mt * 128 + wv * 16;
    const int n0 = nt * 64;

    v8f acc[4] = {};
    gemm_panel(Oi, Wo, ldsB, mrow, n0, wv, hi, l15, acc);

#pragma unroll
    for (int j = 0; j < 4; ++j) {
        const int f = n0 + j * 16 + l15;
#pragma unroll
        for (int v = 0; v < 8; ++v) {
            out[(mrow + v + 8 * hi) * D_MODEL + f] = acc[j][v];
        }
    }
}

// ---------------------------------------------------------------------------
// Launch. Workspace layout (bytes):
//   Xb  @ 0          : 4608*768  bf16 =  7,077,888
//   Wqb @ 7077888    : 2304*768  bf16 =  3,538,944
//   Wob @ 10616832   : 768*768   bf16 =  1,179,648
//   QKb @ 11796480   : 4608*1536 bf16 = 14,155,776
//   Vtb @ 25952256   : 1536*2304 bf16 =  7,077,888
//   Ob  @ 33030144   : 4608*768  bf16 =  7,077,888   (total ~40.1 MB)
// ---------------------------------------------------------------------------
extern "C" void kernel_launch(void* const* d_in, const int* in_sizes, int n_in,
                              void* d_out, int out_size, void* d_ws, size_t ws_size,
                              hipStream_t stream) {
    (void)in_sizes; (void)n_in; (void)out_size; (void)ws_size;
    const float* x  = (const float*)d_in[0];
    const float* wq = (const float*)d_in[1];
    const float* wo = (const float*)d_in[2];
    const int*   ks = (const int*)d_in[3];

    char* ws = (char*)d_ws;
    unsigned short* Xb  = (unsigned short*)(ws + 0);
    unsigned short* Wqb = (unsigned short*)(ws + 7077888);
    unsigned short* Wob = (unsigned short*)(ws + 10616832);
    unsigned short* QKb = (unsigned short*)(ws + 11796480);
    unsigned short* Vtb = (unsigned short*)(ws + 25952256);
    unsigned short* Ob  = (unsigned short*)(ws + 33030144);

    k_cvt<<<3456, 256, 0, stream>>>(x,  Xb,  884736);   // 4608*768 / 4
    k_cvt<<<1728, 256, 0, stream>>>(wq, Wqb, 442368);   // 2304*768 / 4
    k_cvt<<< 576, 256, 0, stream>>>(wo, Wob, 147456);   // 768*768  / 4

    k_gemm_qkv<<<1296, 256, 0, stream>>>(Xb, Wqb, QKb, Vtb);  // 36x36 128x64 tiles
    k_attn    <<< 864, 128, 0, stream>>>(QKb, Vtb, Ob, ks);   // 2*12*144 tasks / 4 waves
    k_gemm_out<<< 432, 256, 0, stream>>>(Ob, Wob, (float*)d_out); // 36x12 tiles
}
